// GraphModule_35390530519350
// MI455X (gfx1250) — matrix operand order repaired and checked by reading, not compiled, not run on hardware
//
#include <hip/hip_runtime.h>
#include <hip/hip_bf16.h>

// ---------------- model dims ----------------
constexpr int V    = 49152;
constexpr int D    = 1536;
constexpr int H    = 16;
constexpr int HD   = 48;
constexpr int F    = 3072;
constexpr int S    = 1024;
constexpr int B    = 2;
constexpr int NTOK = B * S;          // 2048
constexpr int HHD  = H * HD;         // 768
constexpr int QKVN = 3 * HHD;        // 2304

// ---------------- CDNA5 WMMA types ----------------
typedef __attribute__((ext_vector_type(8)))  float        v8f;
typedef __attribute__((ext_vector_type(16))) __bf16       v16bf;
typedef __attribute__((ext_vector_type(8)))  unsigned int v8u;

// ---- async global->LDS helpers (CDNA5 ASYNCcnt path) ----
__device__ inline void async_b128_to_lds(unsigned lds_byte_addr, const unsigned* gsrc) {
  asm volatile("global_load_async_to_lds_b128 %0, %1, off"
               :: "v"(lds_byte_addr), "v"(gsrc) : "memory");
}
__device__ inline void wait_async0() {
  asm volatile("s_wait_asynccnt 0x0" ::: "memory");
}

// A-fragment (16x32 bf16, ISA 7.12.2) from a swizzled LDS tile (16 rows x 16 uints,
// 16B groups XOR-swizzled by (row>>2)&3). lane<16 -> M=lane, kpairs 0..7;
// lane>=16 -> M=lane-16, kpairs 8..15.
__device__ inline v16bf frag_a_swz(const unsigned* base, int lane) {
  const int m  = lane & 15;
  const int kb = (lane >> 4) << 3;          // 0 or 8
  const int s  = (m >> 2) & 3;
  v8u u;
#pragma unroll
  for (int v = 0; v < 8; ++v) {
    const int kp  = kb + v;
    const int col = ((((kp >> 2) ^ s) << 2) | (kp & 3));
    u[v] = base[m * 16 + col];
  }
  return __builtin_bit_cast(v16bf, u);
}

// B-fragment (32x16 bf16) from swizzled LDS tile: lane<16 -> N=lane, kpairs 0..7;
// lane>=16 -> kpairs 8..15.
__device__ inline v16bf frag_b_swz(const unsigned* base, int lane) {
  const int n  = lane & 15;
  const int kb = (lane >> 4) << 3;          // 0 or 8
  const int s  = (n >> 2) & 3;
  v8u u;
#pragma unroll
  for (int v = 0; v < 8; ++v) {
    const int kp  = kb + v;
    const int col = ((((kp >> 2) ^ s) << 2) | (kp & 3));
    u[v] = base[n * 16 + col];
  }
  return __builtin_bit_cast(v16bf, u);
}

// Unswizzled A-fragment with arbitrary row stride (used by the loss kernel).
__device__ inline v16bf frag_a_lin(const unsigned* base, int ldu, int lane) {
  const int m  = lane & 15;
  const int kb = (lane >> 4) << 3;
  v8u u;
#pragma unroll
  for (int v = 0; v < 8; ++v) {
    const int kp = kb + v;
    u[v] = base[m * ldu + kp];
  }
  return __builtin_bit_cast(v16bf, u);
}

// ---------------- elementwise / LN kernels ----------------
__global__ void f32_to_bf16_kernel(const float* __restrict__ in,
                                   __hip_bfloat16* __restrict__ out, long long n) {
  long long i = (long long)blockIdx.x * blockDim.x + threadIdx.x;
  const long long stride = (long long)gridDim.x * blockDim.x;
  for (; i < n; i += stride) out[i] = __float2bfloat16(in[i]);
}

__global__ void embed_kernel(const int* __restrict__ ids,
                             const unsigned char* __restrict__ mask,
                             const float* __restrict__ emb,
                             const float* __restrict__ pos,
                             float* __restrict__ x) {
  const int row = blockIdx.x;                 // b*S+s
  const int s   = row % S;
  const int id  = ids[row];
  const bool m  = mask[row] != 0;
  float* xr       = x   + (size_t)row * D;
  const float* er = emb + (size_t)id  * D;
  const float* pr = pos + (size_t)s   * D;
  for (int d = threadIdx.x; d < D; d += 256)
    xr[d] = m ? 0.f : (er[d] + pr[d]);
}

__global__ void ln_rows_bf16(const float* __restrict__ X,
                             const float* __restrict__ w, const float* __restrict__ b,
                             __hip_bfloat16* __restrict__ Y, int Dlen) {
  const int row = blockIdx.x, tid = threadIdx.x;
  const float* xr = X + (size_t)row * Dlen;
  float s = 0.f, s2 = 0.f;
  for (int d = tid; d < Dlen; d += 256) { float v = xr[d]; s += v; s2 += v * v; }
  __shared__ float rs[256], rq[256];
  rs[tid] = s; rq[tid] = s2; __syncthreads();
  for (int off = 128; off > 0; off >>= 1) {
    if (tid < off) { rs[tid] += rs[tid + off]; rq[tid] += rq[tid + off]; }
    __syncthreads();
  }
  const float mean = rs[0] / Dlen;
  const float var  = rq[0] / Dlen - mean * mean;
  const float rinv = rsqrtf(var + 1e-5f);
  __hip_bfloat16* yr = Y + (size_t)row * Dlen;
  for (int d = tid; d < Dlen; d += 256)
    yr[d] = __float2bfloat16((xr[d] - mean) * rinv * w[d] + b[d]);
}

// per-head LN of q or k slice; writes transposed [B,H,S,HD] f32
__global__ void qk_ln_kernel(const float* __restrict__ qkv,
                             const float* __restrict__ w, const float* __restrict__ b,
                             float* __restrict__ out, int koff) {
  const int idx = blockIdx.x;            // (b*S+s)*H + h
  const int h = idx % H, bs = idx / H;
  const int tid = threadIdx.x;
  const float* xr = qkv + (size_t)bs * QKVN + koff + h * HD;
  const float v = (tid < HD) ? xr[tid] : 0.f;
  __shared__ float rs[64], rq[64];
  rs[tid] = v; rq[tid] = v * v; __syncthreads();
  for (int off = 32; off > 0; off >>= 1) {
    if (tid < off) { rs[tid] += rs[tid + off]; rq[tid] += rq[tid + off]; }
    __syncthreads();
  }
  const float mean = rs[0] / HD;
  const float var  = rq[0] / HD - mean * mean;
  const float rinv = rsqrtf(var + 1e-5f);
  if (tid < HD) {
    const int b_ = bs / S, s_ = bs % S;
    out[(((size_t)b_ * H + h) * S + s_) * HD + tid] = (v - mean) * rinv * w[tid] + b[tid];
  }
}

__global__ void init_count_kernel(const unsigned char* __restrict__ mask,
                                  float* __restrict__ wsf, float* __restrict__ out) {
  __shared__ int rs[256];
  const int tid = threadIdx.x;
  int c = 0;
  for (int i = tid; i < NTOK; i += 256) c += (mask[i] == 0);
  rs[tid] = c; __syncthreads();
  for (int off = 128; off > 0; off >>= 1) {
    if (tid < off) rs[tid] += rs[tid + off];
    __syncthreads();
  }
  if (tid == 0) {
    int cnt = rs[0]; if (cnt < 1) cnt = 1;
    wsf[0] = 1.f / (float)cnt;
    out[0] = 0.f;
  }
}

// ---------------- tiled WMMA GEMM: C = act(A @ B^T + bias) (+residual) ----------------
// A: bf16 [M,K] row-major (as uint pairs), B: bf16 [N,K] row-major.
// 64x64 block tile, 8 waves x two 16x16 accumulators. K staged 32 at a time into
// double-buffered LDS via GLOBAL_LOAD_ASYNC_TO_LDS_B128, overlapped with WMMA.
__global__ __launch_bounds__(256) void gemm_wmma_bf16(
    const unsigned* __restrict__ Au, const unsigned* __restrict__ Bu,
    const float* __restrict__ bias, const float* __restrict__ residual,
    float* __restrict__ Cf, __hip_bfloat16* __restrict__ Cb,
    int M, int N, int K, int act_gelu) {
  __shared__ __align__(16) unsigned As[2][64 * 16];
  __shared__ __align__(16) unsigned Bs[2][64 * 16];
  const int tid  = threadIdx.x;
  const int lane = tid & 31, w = tid >> 5;
  const int m0 = blockIdx.y * 64, n0 = blockIdx.x * 64;
  const int Ku = K >> 1;                 // uints per row
  const int sm = w & 3, sn0 = (w >> 2) * 2;

  // async staging: each thread copies one 16B group of A and one of B.
  const unsigned abase = (unsigned)(uintptr_t)&As[0][0];
  const unsigned bbase = (unsigned)(uintptr_t)&Bs[0][0];
  const int r  = tid >> 2;                       // tile row 0..63
  const int cg = tid & 3;                        // logical 16B group
  const int pg = cg ^ ((r >> 2) & 3);            // XOR-swizzled group
  const unsigned lds_off = (unsigned)((r * 16 + pg * 4) * 4);

  v8f acc0 = {}; v8f acc1 = {};
  const int nsteps = Ku / 16;                    // 32 K-elements per step

  // prologue: fill buffer 0
  async_b128_to_lds(abase + lds_off, Au + (size_t)(m0 + r) * Ku + 0 + cg * 4);
  async_b128_to_lds(bbase + lds_off, Bu + (size_t)(n0 + r) * Ku + 0 + cg * 4);
  wait_async0();
  __syncthreads();

  for (int step = 0; step < nsteps; ++step) {
    const int cur = step & 1;
    if (step + 1 < nsteps) {
      const int ks = (step + 1) * 16;
      const unsigned boff = (unsigned)((cur ^ 1) * 64 * 16 * 4);
      async_b128_to_lds(abase + boff + lds_off, Au + (size_t)(m0 + r) * Ku + ks + cg * 4);
      async_b128_to_lds(bbase + boff + lds_off, Bu + (size_t)(n0 + r) * Ku + ks + cg * 4);
    }
    const v16bf a  = frag_a_swz(&As[cur][sm * 16 * 16], lane);
    const v16bf b0 = frag_b_swz(&Bs[cur][sn0 * 16 * 16], lane);
    const v16bf b1 = frag_b_swz(&Bs[cur][(sn0 + 1) * 16 * 16], lane);
    acc0 = __builtin_amdgcn_wmma_f32_16x16x32_bf16(false, a, false, b0, (short)0, acc0, false, false);
    acc1 = __builtin_amdgcn_wmma_f32_16x16x32_bf16(false, a, false, b1, (short)0, acc1, false, false);
    wait_async0();
    __syncthreads();
  }

  const int mbase = m0 + sm * 16 + (lane >> 4) * 8;
  const int nc0   = n0 + sn0 * 16 + (lane & 15);
  const int nc1   = nc0 + 16;
#pragma unroll
  for (int rr = 0; rr < 8; ++rr) {
    const int m = mbase + rr;
#pragma unroll
    for (int t = 0; t < 2; ++t) {
      const int n = (t == 0) ? nc0 : nc1;
      float v = (t == 0) ? acc0[rr] : acc1[rr];
      if (bias)      v += bias[n];
      if (act_gelu)  v = 0.5f * v * (1.f + erff(v * 0.70710678118654752f));
      if (residual)  v += residual[(size_t)m * N + n];
      if (Cf) Cf[(size_t)m * N + n] = v;
      if (Cb) Cb[(size_t)m * N + n] = __float2bfloat16(v);
    }
  }
}

// ---------------- causal attention (online softmax, one thread per query row) ----------------
__global__ __launch_bounds__(256) void attn_kernel(
    const float* __restrict__ q2, const float* __restrict__ k2,
    const float* __restrict__ qkv, __hip_bfloat16* __restrict__ aout) {
  const int bh = blockIdx.x;                  // b*H + h
  const int b  = bh >> 4, h = bh & 15;
  const int q  = blockIdx.y * 256 + threadIdx.x;
  const float4* qp = (const float4*)(q2 + ((size_t)bh * S + q) * HD);
  float4 qv[12];
#pragma unroll
  for (int i = 0; i < 12; ++i) qv[i] = qp[i];
  float mx = -3.0e38f, sm = 0.f;
  float acc[HD];
#pragma unroll
  for (int d = 0; d < HD; ++d) acc[d] = 0.f;
  const float scale = 0.14433756729740643f;   // 1/sqrt(48)
  for (int k = 0; k <= q; ++k) {
    const float4* kp = (const float4*)(k2 + ((size_t)bh * S + k) * HD);
    float dot = 0.f;
#pragma unroll
    for (int i = 0; i < 12; ++i) {
      const float4 kv = kp[i];
      dot += qv[i].x * kv.x + qv[i].y * kv.y + qv[i].z * kv.z + qv[i].w * kv.w;
    }
    const float s  = dot * scale;
    const float nm = fmaxf(mx, s);
    const float c  = __expf(mx - nm);
    const float p  = __expf(s - nm);
    sm = sm * c + p;
    const float4* vp = (const float4*)(qkv + (size_t)(b * S + k) * QKVN + 2 * HHD + h * HD);
#pragma unroll
    for (int i = 0; i < 12; ++i) {
      const float4 vv = vp[i];
      acc[4 * i + 0] = acc[4 * i + 0] * c + p * vv.x;
      acc[4 * i + 1] = acc[4 * i + 1] * c + p * vv.y;
      acc[4 * i + 2] = acc[4 * i + 2] * c + p * vv.z;
      acc[4 * i + 3] = acc[4 * i + 3] * c + p * vv.w;
    }
    mx = nm;
  }
  const float inv = 1.f / sm;
  __hip_bfloat16* ar = aout + (size_t)(b * S + q) * HHD + h * HD;
#pragma unroll
  for (int d = 0; d < HD; ++d) ar[d] = __float2bfloat16(acc[d] * inv);
}

// ---------------- fused logits GEMM + logsumexp + target gather + NLL ----------------
// Block = 16 token rows; h3 tile (48KB) async-copied to LDS; 8 waves sweep all 3072
// vocab tiles with WMMA; bf16 emb (151MB) is L2-resident on MI455X (192MB L2).
__global__ __launch_bounds__(256) void loss_kernel(
    const unsigned* __restrict__ Hu,          // h3 bf16 as uint [NTOK][D/2]
    const unsigned* __restrict__ Eu,          // emb bf16 as uint [V][D/2]
    const int* __restrict__ targets,
    const unsigned char* __restrict__ mask,
    const float* __restrict__ wsf,            // wsf[0] = 1/count
    float* __restrict__ out) {
  __shared__ __align__(16) unsigned Ht[16 * 768];
  __shared__ float redm[8][2][8], reds[8][2][8], redt[8][2][8];
  const int tid = threadIdx.x, lane = tid & 31, w = tid >> 5;
  const int row0 = blockIdx.x * 16;

  // async-stage the contiguous 16x1536 bf16 tile (3072 x 16B)
  const unsigned hbase = (unsigned)(uintptr_t)&Ht[0];
#pragma unroll
  for (int j = 0; j < 12; ++j) {
    const int q = tid + j * 256;               // uint4 index 0..3071
    async_b128_to_lds(hbase + (unsigned)(q * 16),
                      Hu + (size_t)row0 * 768 + (size_t)q * 4);
  }
  wait_async0();
  __syncthreads();

  const int half = lane >> 4;
  float mx[8], sm[8], tg[8];
#pragma unroll
  for (int r = 0; r < 8; ++r) { mx[r] = -3.0e38f; sm[r] = 0.f; tg[r] = -3.0e38f; }

  for (int tile = w; tile < V / 16; tile += 8) {
    const int vb = tile * 16;
    if (tile + 8 < V / 16)
      __builtin_prefetch(Eu + (size_t)((tile + 8) * 16 + (lane & 15)) * 768, 0, 1);
    v8f acc = {};
    const unsigned* Brow = Eu + (size_t)(vb + (lane & 15)) * 768;
    const int kb = (lane >> 4) << 3;
#pragma unroll 1
    for (int ks = 0; ks < 48; ++ks) {                  // 48 * K32 = 1536
      const v16bf a = frag_a_lin(&Ht[ks * 16], 768, lane);
      const uint4* B4 = (const uint4*)(Brow + ks * 16 + kb);   // 16B aligned
      const uint4 p0 = B4[0];
      const uint4 p1 = B4[1];
      v8u ub;
      ub[0] = p0.x; ub[1] = p0.y; ub[2] = p0.z; ub[3] = p0.w;
      ub[4] = p1.x; ub[5] = p1.y; ub[6] = p1.z; ub[7] = p1.w;
      const v16bf bfr = __builtin_bit_cast(v16bf, ub);
      acc = __builtin_amdgcn_wmma_f32_16x16x32_bf16(false, a, false, bfr, (short)0, acc, false, false);
    }
    const int vcol = vb + (lane & 15);
#pragma unroll
    for (int r = 0; r < 8; ++r) {
      const float val = acc[r];
      const int gr = row0 + half * 8 + r;
      if (targets[gr] == vcol) tg[r] = val;
      const float nm = fmaxf(mx[r], val);
      sm[r] = sm[r] * __expf(mx[r] - nm) + __expf(val - nm);
      mx[r] = nm;
    }
  }

  // merge 16 column-lanes within each half via shuffles
#pragma unroll
  for (int off = 1; off < 16; off <<= 1) {
#pragma unroll
    for (int r = 0; r < 8; ++r) {
      const float omx = __shfl_xor(mx[r], off, 32);
      const float osm = __shfl_xor(sm[r], off, 32);
      const float otg = __shfl_xor(tg[r], off, 32);
      const float nm = fmaxf(mx[r], omx);
      sm[r] = sm[r] * __expf(mx[r] - nm) + osm * __expf(omx - nm);
      mx[r] = nm;
      tg[r] = fmaxf(tg[r], otg);
    }
  }
  if ((lane & 15) == 0) {
#pragma unroll
    for (int r = 0; r < 8; ++r) {
      redm[w][half][r] = mx[r]; reds[w][half][r] = sm[r]; redt[w][half][r] = tg[r];
    }
  }
  __syncthreads();
  if (tid < 16) {
    const int m = tid, hh = m >> 3, r = m & 7;
    float fm = -3.0e38f, fs = 0.f, ft = -3.0e38f;
    for (int ww = 0; ww < 8; ++ww) {
      const float omx = redm[ww][hh][r], osm = reds[ww][hh][r], otg = redt[ww][hh][r];
      const float nm = fmaxf(fm, omx);
      fs = fs * __expf(fm - nm) + osm * __expf(omx - nm);
      fm = nm;
      ft = fmaxf(ft, otg);
    }
    const int gr = row0 + m;
    const float logz = fm + __logf(fs);
    const float nll  = mask[gr] ? 0.f : (logz - ft);
    atomicAdd(out, nll * wsf[0]);
  }
}

// ---------------- host launcher ----------------
extern "C" void kernel_launch(void* const* d_in, const int* in_sizes, int n_in,
                              void* d_out, int out_size, void* d_ws, size_t ws_size,
                              hipStream_t stream) {
  (void)in_sizes; (void)n_in; (void)out_size; (void)ws_size;
  const int*           ids     = (const int*)d_in[0];
  const int*           targets = (const int*)d_in[1];
  const unsigned char* mask    = (const unsigned char*)d_in[2];
  const float* emb   = (const float*)d_in[3];
  const float* pos   = (const float*)d_in[4];
  const float* ln1w  = (const float*)d_in[5];
  const float* ln1b  = (const float*)d_in[6];
  const float* qkvw  = (const float*)d_in[7];
  const float* qkvb  = (const float*)d_in[8];
  const float* qlnw  = (const float*)d_in[9];
  const float* qlnb  = (const float*)d_in[10];
  const float* klnw  = (const float*)d_in[11];
  const float* klnb  = (const float*)d_in[12];
  const float* outw  = (const float*)d_in[13];
  const float* outb  = (const float*)d_in[14];
  const float* ln2w  = (const float*)d_in[15];
  const float* ln2b  = (const float*)d_in[16];
  const float* fc1w  = (const float*)d_in[17];
  const float* fc1b  = (const float*)d_in[18];
  const float* fc2w  = (const float*)d_in[19];
  const float* fc2b  = (const float*)d_in[20];
  const float* lnfw  = (const float*)d_in[21];
  const float* lnfb  = (const float*)d_in[22];

  char* ws = (char*)d_ws;
  size_t off = 0;
  auto alloc = [&](size_t bytes) -> void* {
    off = (off + 255) & ~(size_t)255;
    void* p = ws + off;
    off += bytes;
    return p;
  };

  float*          wsf     = (float*)alloc(256);
  __hip_bfloat16* emb_bf  = (__hip_bfloat16*)alloc((size_t)V * D * 2);
  __hip_bfloat16* qkvw_bf = (__hip_bfloat16*)alloc((size_t)QKVN * D * 2);
  __hip_bfloat16* outw_bf = (__hip_bfloat16*)alloc((size_t)D * HHD * 2);
  __hip_bfloat16* fc1w_bf = (__hip_bfloat16*)alloc((size_t)F * D * 2);
  __hip_bfloat16* fc2w_bf = (__hip_bfloat16*)alloc((size_t)D * F * 2);
  float*          x       = (float*)alloc((size_t)NTOK * D * 4);
  __hip_bfloat16* h_bf    = (__hip_bfloat16*)alloc((size_t)NTOK * D * 2);
  float*          qkv     = (float*)alloc((size_t)NTOK * QKVN * 4);
  float*          q2      = (float*)alloc((size_t)NTOK * HHD * 4);
  float*          k2      = (float*)alloc((size_t)NTOK * HHD * 4);
  __hip_bfloat16* a_bf    = (__hip_bfloat16*)alloc((size_t)NTOK * HHD * 2);
  float*          x2      = (float*)alloc((size_t)NTOK * D * 4);
  __hip_bfloat16* h2_bf   = (__hip_bfloat16*)alloc((size_t)NTOK * D * 2);
  __hip_bfloat16* m_bf    = (__hip_bfloat16*)alloc((size_t)NTOK * F * 2);
  float*          x3      = (float*)alloc((size_t)NTOK * D * 4);
  __hip_bfloat16* h3_bf   = (__hip_bfloat16*)alloc((size_t)NTOK * D * 2);

  // 0) count + zero output
  init_count_kernel<<<1, 256, 0, stream>>>(mask, wsf, (float*)d_out);

  // 1) f32 -> bf16 weight conversions (emb_bf becomes L2-resident for the loss pass)
  f32_to_bf16_kernel<<<4096, 256, 0, stream>>>(emb,  emb_bf,  (long long)V * D);
  f32_to_bf16_kernel<<<1024, 256, 0, stream>>>(qkvw, qkvw_bf, (long long)QKVN * D);
  f32_to_bf16_kernel<<<512,  256, 0, stream>>>(outw, outw_bf, (long long)D * HHD);
  f32_to_bf16_kernel<<<1024, 256, 0, stream>>>(fc1w, fc1w_bf, (long long)F * D);
  f32_to_bf16_kernel<<<1024, 256, 0, stream>>>(fc2w, fc2w_bf, (long long)D * F);

  // 2) embedding + positions
  embed_kernel<<<NTOK, 256, 0, stream>>>(ids, mask, emb, pos, x);

  // 3) LN1 -> bf16
  ln_rows_bf16<<<NTOK, 256, 0, stream>>>(x, ln1w, ln1b, h_bf, D);

  // 4) QKV GEMM  [2048,1536] @ [2304,1536]^T
  gemm_wmma_bf16<<<dim3(QKVN / 64, NTOK / 64), 256, 0, stream>>>(
      (const unsigned*)h_bf, (const unsigned*)qkvw_bf, qkvb, nullptr,
      qkv, nullptr, NTOK, QKVN, D, 0);

  // 5) per-head QK layernorm -> [B,H,S,HD]
  qk_ln_kernel<<<NTOK * H, 64, 0, stream>>>(qkv, qlnw, qlnb, q2, 0);
  qk_ln_kernel<<<NTOK * H, 64, 0, stream>>>(qkv, klnw, klnb, k2, HHD);

  // 6) causal attention -> a_bf [2048,768] bf16
  attn_kernel<<<dim3(B * H, S / 256), 256, 0, stream>>>(q2, k2, qkv, a_bf);

  // 7) out projection + residual: x2 = x + a @ out_w^T + out_b
  gemm_wmma_bf16<<<dim3(D / 64, NTOK / 64), 256, 0, stream>>>(
      (const unsigned*)a_bf, (const unsigned*)outw_bf, outb, x,
      x2, nullptr, NTOK, D, HHD, 0);

  // 8) LN2 -> bf16
  ln_rows_bf16<<<NTOK, 256, 0, stream>>>(x2, ln2w, ln2b, h2_bf, D);

  // 9) FC1 + exact GELU -> bf16
  gemm_wmma_bf16<<<dim3(F / 64, NTOK / 64), 256, 0, stream>>>(
      (const unsigned*)h2_bf, (const unsigned*)fc1w_bf, fc1b, nullptr,
      nullptr, m_bf, NTOK, F, D, 1);

  // 10) FC2 + residual: x3 = x2 + gelu @ fc2_w^T + fc2_b
  gemm_wmma_bf16<<<dim3(D / 64, NTOK / 64), 256, 0, stream>>>(
      (const unsigned*)m_bf, (const unsigned*)fc2w_bf, fc2b, x2,
      x3, nullptr, NTOK, D, F, 0);

  // 11) final LN -> bf16
  ln_rows_bf16<<<NTOK, 256, 0, stream>>>(x3, lnfw, lnfb, h3_bf, D);

  // 12) fused logits + logsumexp + NLL
  loss_kernel<<<NTOK / 16, 256, 0, stream>>>(
      (const unsigned*)h3_bf, (const unsigned*)emb_bf, targets, mask, wsf,
      (float*)d_out);
}
